// TMHA_26010321945088
// MI455X (gfx1250) — compile-verified, matrix operand order checked by reading, not run on hardware
//
#include <hip/hip_runtime.h>
#include <hip/hip_bf16.h>

// ---------------------------------------------------------------------------
// Types for CDNA5 WMMA (wave32)
// ---------------------------------------------------------------------------
typedef __attribute__((ext_vector_type(16))) __bf16 v16bf;
typedef __attribute__((ext_vector_type(8)))  __bf16 v8bf;
typedef __attribute__((ext_vector_type(4)))  __bf16 v4bf;
typedef __attribute__((ext_vector_type(8)))  float  v8f;
typedef __attribute__((ext_vector_type(4)))  float  v4f;

__device__ __forceinline__ __bf16 f2bf(float f) {
  unsigned u = __builtin_bit_cast(unsigned, f);
  unsigned r = u + 0x7FFFu + ((u >> 16) & 1u);   // round-to-nearest-even
  unsigned short h = (unsigned short)(r >> 16);
  return __builtin_bit_cast(__bf16, h);
}

// CDNA5 async copy: 16B global -> LDS per lane, tracked with ASYNCcnt.
// LDS destination address = low 32 bits of the generic shared pointer.
__device__ __forceinline__ void async_cp16(void* lds, const void* gptr) {
  unsigned l = (unsigned)(uintptr_t)lds;
  asm volatile("global_load_async_to_lds_b128 %0, %1, off"
               :: "v"(l), "v"(gptr) : "memory");
}
__device__ __forceinline__ void async_wait0() {
  asm volatile("s_wait_asynccnt 0x0" ::: "memory");
}
__device__ __forceinline__ void async_wait4() {   // counted wait: <=4 outstanding
  asm volatile("s_wait_asynccnt 0x4" ::: "memory");
}

// Load one 16x32 bf16 WMMA operand fragment from LDS laid out as
// [striped_dim][K] row-major, row pointer already resolved for this lane.
// elems 0..7 -> K=kbase..kbase+7, elems 8..15 -> K=16+kbase.., kbase=8*(lane>=16)
__device__ __forceinline__ v16bf frag_ld(const __bf16* p) {
  v8bf lo = *(const v8bf*)(p);
  v8bf hi = *(const v8bf*)(p + 16);
  return __builtin_shufflevector(lo, hi, 0,1,2,3,4,5,6,7,8,9,10,11,12,13,14,15);
}

__device__ __forceinline__ v8f wmma_bf16(v16bf a, v16bf b, v8f c) {
  return __builtin_amdgcn_wmma_f32_16x16x32_bf16(false, a, false, b, (short)0, c,
                                                 false, false);
}

// ---------------------------------------------------------------------------
// Kernel 0: fp32 -> bf16 convert (4 elems / thread)
// ---------------------------------------------------------------------------
__global__ __launch_bounds__(256) void cvt_kernel(const float* __restrict__ in,
                                                  __bf16* __restrict__ out, int n4) {
  int i = blockIdx.x * 256 + threadIdx.x;
  if (i < n4) {
    v4f v = *(const v4f*)(in + (size_t)i * 4);
    v4bf o;
    o[0] = f2bf(v[0]); o[1] = f2bf(v[1]); o[2] = f2bf(v[2]); o[3] = f2bf(v[3]);
    *(v4bf*)(out + (size_t)i * 4) = o;
  }
}

// ---------------------------------------------------------------------------
// Kernel 0b: fp32 [Kd,Nd] -> bf16 transposed [Nd,Kd] (weights, done once)
// 64x64 tile per block via LDS.
// ---------------------------------------------------------------------------
__global__ __launch_bounds__(256) void cvtT_kernel(const float* __restrict__ in,
                                                   __bf16* __restrict__ outT,
                                                   int Kd, int Nd) {
  __shared__ float tile[64 * 65];
  const int k0 = blockIdx.x * 64, n0 = blockIdx.y * 64;
  const int tid = threadIdx.x;
  const int r = tid >> 2, c0 = (tid & 3) * 16;
  #pragma unroll
  for (int q = 0; q < 4; ++q) {
    v4f v = *(const v4f*)(in + (size_t)(k0 + r) * Nd + n0 + c0 + q * 4);
    tile[r * 65 + c0 + q * 4 + 0] = v[0];
    tile[r * 65 + c0 + q * 4 + 1] = v[1];
    tile[r * 65 + c0 + q * 4 + 2] = v[2];
    tile[r * 65 + c0 + q * 4 + 3] = v[3];
  }
  __syncthreads();
  const int n = tid >> 2, kc = (tid & 3) * 16;
  v8bf o0, o1;
  #pragma unroll
  for (int j = 0; j < 8; ++j) {
    o0[j] = f2bf(tile[(kc + j) * 65 + n]);
    o1[j] = f2bf(tile[(kc + 8 + j) * 65 + n]);
  }
  __bf16* dst = outT + (size_t)(n0 + n) * Kd + k0 + kc;
  *(v8bf*)dst = o0;
  *(v8bf*)(dst + 8) = o1;
}

// ---------------------------------------------------------------------------
// Kernel 0c: bf16 per-batch transpose v[b][s][c] -> vT[b][c][s]  (1024x1024)
// ---------------------------------------------------------------------------
__global__ __launch_bounds__(256) void transpose_v_kernel(
    const __bf16* __restrict__ vbf, __bf16* __restrict__ vT) {
  __shared__ __align__(16) __bf16 tile[64 * 72];
  const int s0 = blockIdx.x * 64, c0 = blockIdx.y * 64, b = blockIdx.z;
  const int tid = threadIdx.x;
  const int r = tid >> 2, cc = (tid & 3) * 16;
  const __bf16* src = vbf + ((size_t)b * 1024 + s0 + r) * 1024 + c0 + cc;
  *(v8bf*)&tile[r * 72 + cc]     = *(const v8bf*)src;
  *(v8bf*)&tile[r * 72 + cc + 8] = *(const v8bf*)(src + 8);
  __syncthreads();
  const int c = tid >> 2, ss = (tid & 3) * 16;
  v8bf o0, o1;
  #pragma unroll
  for (int j = 0; j < 8; ++j) {
    o0[j] = tile[(ss + j) * 72 + c];
    o1[j] = tile[(ss + 8 + j) * 72 + c];
  }
  __bf16* dst = vT + ((size_t)b * 1024 + c0 + c) * 1024 + s0 + ss;
  *(v8bf*)dst = o0;
  *(v8bf*)(dst + 8) = o1;
}

// ---------------------------------------------------------------------------
// Kernel 1: bf16 GEMM  C[M,N](f32) = A[M,K](bf16) @ BT[N,K](bf16, pre-transposed)
// Block tile 64x128, K-slab 64, double-buffered async-to-LDS DMA for A and B.
// 8 waves (2x4), 32x32 per wave, 8 WMMA per wave per slab.
// ---------------------------------------------------------------------------
__global__ __launch_bounds__(256) void gemm_bf16_kernel(
    const __bf16* __restrict__ A, const __bf16* __restrict__ BT,
    float* __restrict__ C, int M, int N, int K) {
  constexpr int BM = 64, BN = 128, BK = 64;
  constexpr int LDA = BK + 8;         // 72 elems; 144B rows, 16B-aligned chunks
  constexpr int LDB = BK + 8;
  constexpr int ASZ = BM * LDA;       // one A buffer
  constexpr int BSZ = BN * LDB;       // one B buffer
  __shared__ __align__(16) __bf16 As[2 * ASZ];
  __shared__ __align__(16) __bf16 Bt[2 * BSZ];

  const int m0 = blockIdx.x * BM;
  const int n0 = blockIdx.y * BN;
  const int tid = threadIdx.x;
  const int lane = tid & 31;
  const int w = tid >> 5;
  const int wm = w >> 2;              // 0..1
  const int wn = w & 3;               // 0..3
  const int l15 = lane & 15;
  const int kb8 = (lane >> 4) * 8;    // fragment K base

  // issue one slab's DMA (6 async ops / thread) into buffer pb
  auto issue = [&](int k0, int pb) {
    __bf16* Asb = As + pb * ASZ;
    __bf16* Btb = Bt + pb * BSZ;
    {
      int idx = tid * 16;
      int row = idx >> 6, col = idx & 63;
      const __bf16* src = A + (size_t)(m0 + row) * K + k0 + col;
      async_cp16(Asb + row * LDA + col,     src);
      async_cp16(Asb + row * LDA + col + 8, src + 8);
    }
    #pragma unroll
    for (int p = 0; p < 4; ++p) {
      int c = tid + p * 256;          // 1024 chunks of 8
      int n = c >> 3, k8 = (c & 7) * 8;
      async_cp16(Btb + n * LDB + k8, BT + (size_t)(n0 + n) * K + k0 + k8);
    }
  };

  v8f acc[2][2] = {};
  const int nslab = K / BK;
  issue(0, 0);
  for (int it = 0; it < nslab; ++it) {
    async_wait0();                    // slab `it` DMA complete (own ops)
    __syncthreads();                  // visible to all; prior compute closed
    if (it + 1 < nslab) issue((it + 1) * BK, (it + 1) & 1);
    const __bf16* Asb = As + (it & 1) * ASZ;
    const __bf16* Btb = Bt + (it & 1) * BSZ;

    #pragma unroll
    for (int ks = 0; ks < 2; ++ks) {  // two 32-deep K steps inside slab
      v16bf bfr[2];
      #pragma unroll
      for (int j = 0; j < 2; ++j)
        bfr[j] = frag_ld(Btb + (wn * 32 + j * 16 + l15) * LDB + ks * 32 + kb8);
      #pragma unroll
      for (int i = 0; i < 2; ++i) {
        v16bf a = frag_ld(Asb + (wm * 32 + i * 16 + l15) * LDA + ks * 32 + kb8);
        #pragma unroll
        for (int j = 0; j < 2; ++j)
          acc[i][j] = wmma_bf16(a, bfr[j], acc[i][j]);
      }
    }
  }

  const int rhi = kb8;  // accumulator row offset = 8*(lane>=16)
  #pragma unroll
  for (int i = 0; i < 2; ++i)
    #pragma unroll
    for (int j = 0; j < 2; ++j) {
      int mb = m0 + wm * 32 + i * 16 + rhi;
      int n = n0 + wn * 32 + j * 16 + l15;
      #pragma unroll
      for (int r = 0; r < 8; ++r)
        C[(size_t)(mb + r) * N + n] = acc[i][j][r];
    }
}

// ---------------------------------------------------------------------------
// Kernel 2: masked rotation q<-Rq, k<-R^T k, plus bf16 repack of q,k,v
// One block per (b,s). t row layout: [q 0..511 | k 512..1023 | v 1024..2047]
// ---------------------------------------------------------------------------
__global__ __launch_bounds__(256) void rot_kernel(
    const float* __restrict__ t, const float* __restrict__ rot,
    const unsigned char* __restrict__ mask,
    __bf16* __restrict__ qbf, __bf16* __restrict__ kbf,
    __bf16* __restrict__ vbf) {
  __shared__ __align__(16) float Rl[4096];
  __shared__ float ql[512], kl[512];

  const int bs = blockIdx.x;
  const int tid = threadIdx.x;
  const float* trow = t + (size_t)bs * 2048;
  const bool m = mask[bs] != 0;

  ql[tid]       = trow[tid];
  ql[tid + 256] = trow[tid + 256];
  kl[tid]       = trow[512 + tid];
  kl[tid + 256] = trow[768 + tid];
  if (m) {
    const float* Rg = rot + (size_t)bs * 4096;
    #pragma unroll
    for (int p = 0; p < 4; ++p) {
      int i4 = tid * 4 + p * 1024;
      async_cp16(Rl + i4, Rg + i4);
    }
  }
  async_wait0();
  __syncthreads();

  #pragma unroll
  for (int p = 0; p < 4; ++p) {
    int i = tid + p * 256;
    vbf[(size_t)bs * 1024 + i] = f2bf(trow[1024 + i]);
  }

  #pragma unroll
  for (int rep = 0; rep < 2; ++rep) {
    int oid = tid + rep * 256;        // 0..511
    int hh = oid >> 6, ii = oid & 63;
    float qo, ko;
    if (m) {
      float aq = 0.f, ak = 0.f;
      for (int j = 0; j < 64; ++j) {
        aq += Rl[ii * 64 + j] * ql[hh * 64 + j];  // q'_i = sum_j R[i][j] q[j]
        ak += Rl[j * 64 + ii] * kl[hh * 64 + j];  // k'_i = sum_j R[j][i] k[j]
      }
      qo = aq; ko = ak;
    } else {
      qo = ql[hh * 64 + ii];
      ko = kl[hh * 64 + ii];
    }
    qbf[(size_t)bs * 512 + oid] = f2bf(qo);
    kbf[(size_t)bs * 512 + oid] = f2bf(ko);
  }
}

// ---------------------------------------------------------------------------
// Kernel 3: Taylor attention per (b,h, 64-query tile)
//   s = (Q K^T)/8 ; sc = (1 + s + s^2/2)*mask ; denom = sum_k sc
//   O = (sc @ V)/max(denom,1e-6), written bf16 at [b,q, h*128+d]
// All staging via async-to-LDS; V pre-transposed to [b][c][s], so the V tile
// DMA overlaps the score phase (counted s_wait_asynccnt 4 releases K early).
// ---------------------------------------------------------------------------
__global__ __launch_bounds__(256) void attn_kernel(
    const __bf16* __restrict__ Q, const __bf16* __restrict__ Kb,
    const __bf16* __restrict__ vT, const unsigned char* __restrict__ mask,
    __bf16* __restrict__ Ob) {
  constexpr int S = 1024;
  constexpr int LQ = 72, LK = 72, LS = 72, LV = 72;
  __shared__ __align__(16) __bf16 Qs[64 * LQ];
  __shared__ __align__(16) __bf16 Ks[64 * LK];
  __shared__ __align__(16) __bf16 Ss[64 * LS];     // bf16 score tile [q][key]
  __shared__ __align__(16) __bf16 Vt[128 * LV];    // V^T tile: [vcol][key]
  __shared__ float denom[64];

  const int b = blockIdx.x >> 3, h = blockIdx.x & 7;
  const int q0 = blockIdx.y * 64;
  const int tid = threadIdx.x;
  const int lane = tid & 31;
  const int w = tid >> 5;
  const int l15 = lane & 15;
  const int kb8 = (lane >> 4) * 8;
  const int wm = w >> 1;   // 0..3 : 16-row block (scores & output)
  const int wn = w & 1;    // 0..1 : half of keys (scores) / vdims (output)

  // Q tile 64x64 (async; drained by first counted wait inside the loop)
  {
    int idx = tid * 16;
    int row = idx >> 6, col = idx & 63;
    const __bf16* src = Q + ((size_t)(b * S + q0 + row) * 8 + h) * 64 + col;
    async_cp16(Qs + row * LQ + col,     src);
    async_cp16(Qs + row * LQ + col + 8, src + 8);
  }
  if (tid < 64) denom[tid] = 0.f;

  v8f acc[4] = {};
  const float scale = 0.125f;   // 64^-0.5

  for (int kblk = 0; kblk < S / 64; ++kblk) {
    const int k0 = kblk * 64;
    __syncthreads();   // close previous O-phase reads (Vt, Ss); denom init
    // K tile 64x64 (2 async ops)
    {
      int idx = tid * 16;
      int row = idx >> 6, col = idx & 63;
      const __bf16* src = Kb + ((size_t)(b * S + k0 + row) * 8 + h) * 64 + col;
      async_cp16(Ks + row * LK + col,     src);
      async_cp16(Ks + row * LK + col + 8, src + 8);
    }
    // V^T tile 128x64 (4 async ops), contiguous from vT[b][h*128+d][s]
    #pragma unroll
    for (int p = 0; p < 4; ++p) {
      int c = tid + p * 256;            // 1024 chunks of 8
      int d = c >> 3, k8 = (c & 7) * 8;
      const __bf16* src = vT + ((size_t)b * 1024 + h * 128 + d) * 1024 + k0 + k8;
      async_cp16(Vt + d * LV + k8, src);
    }
    async_wait4();     // in-order completion: Q (first iter) + K tile done
    __syncthreads();

    // scores: wave handles subtiles (wm, sn=wn*2+jj)
    {
      v16bf aq0 = frag_ld(Qs + (wm * 16 + l15) * LQ + kb8);
      v16bf aq1 = frag_ld(Qs + (wm * 16 + l15) * LQ + 32 + kb8);
      #pragma unroll
      for (int jj = 0; jj < 2; ++jj) {
        int sn = wn * 2 + jj;
        v16bf bk0 = frag_ld(Ks + (sn * 16 + l15) * LK + kb8);
        v16bf bk1 = frag_ld(Ks + (sn * 16 + l15) * LK + 32 + kb8);
        v8f sc = {};
        sc = wmma_bf16(aq0, bk0, sc);
        sc = wmma_bf16(aq1, bk1, sc);
        int ncol = sn * 16 + l15;
        float mk = mask[b * S + k0 + ncol] ? 1.0f : 0.0f;
        int mrow0 = wm * 16 + kb8;
        #pragma unroll
        for (int r = 0; r < 8; ++r) {
          float sv = sc[r] * scale;
          float val = (1.0f + sv + 0.5f * sv * sv) * mk;
          atomicAdd(&denom[mrow0 + r], val);     // ds_add_f32
          Ss[(mrow0 + r) * LS + ncol] = f2bf(val);
        }
      }
    }
    async_wait0();     // V^T tile done (overlapped with score phase)
    __syncthreads();   // also closes Ss writes

    // O += Ss @ V : wave tiles m = wm*16, n = wn*64 + j*16
    {
      v16bf as0 = frag_ld(Ss + (wm * 16 + l15) * LS + kb8);
      v16bf as1 = frag_ld(Ss + (wm * 16 + l15) * LS + 32 + kb8);
      #pragma unroll
      for (int j = 0; j < 4; ++j) {
        int n = wn * 64 + j * 16 + l15;
        v16bf bv0 = frag_ld(Vt + n * LV + kb8);
        v16bf bv1 = frag_ld(Vt + n * LV + 32 + kb8);
        acc[j] = wmma_bf16(as0, bv0, acc[j]);
        acc[j] = wmma_bf16(as1, bv1, acc[j]);
      }
    }
  }
  __syncthreads();

  // normalize + store bf16 attention output [b, q, h*128 + d]
  {
    int mrow0 = wm * 16 + kb8;
    #pragma unroll
    for (int j = 0; j < 4; ++j) {
      int ncol = h * 128 + wn * 64 + j * 16 + l15;
      #pragma unroll
      for (int r = 0; r < 8; ++r) {
        float d = denom[mrow0 + r];
        d = d > 1e-6f ? d : 1e-6f;
        Ob[(size_t)(b * S + q0 + mrow0 + r) * 1024 + ncol] = f2bf(acc[j][r] / d);
      }
    }
  }
}

// ---------------------------------------------------------------------------
// Host-side launch
// ---------------------------------------------------------------------------
extern "C" void kernel_launch(void* const* d_in, const int* in_sizes, int n_in,
                              void* d_out, int out_size, void* d_ws, size_t ws_size,
                              hipStream_t stream) {
  const float*         x    = (const float*)d_in[0];
  const unsigned char* mask = (const unsigned char*)d_in[1];
  const float*         rot  = (const float*)d_in[2];
  const float*         w_t  = (const float*)d_in[3];
  const float*         w_o  = (const float*)d_in[4];
  float*               out  = (float*)d_out;

  char* ws = (char*)d_ws;
  size_t off = 0;
  auto take = [&](size_t bytes) {
    char* p = ws + off;
    off += (bytes + 255) & ~(size_t)255;
    return p;
  };
  __bf16* xbf  = (__bf16*)take(4194304ull * 2);  // x bf16 [4096,1024]
  __bf16* wtT  = (__bf16*)take(2097152ull * 2);  // w_t^T bf16 [2048,1024]
  __bf16* woT  = (__bf16*)take(1048576ull * 2);  // w_o^T bf16 [1024,1024]
  float*  t    = (float*) take(8388608ull * 4);  // t f32 [4096,2048]
  __bf16* qbf  = (__bf16*)take(2097152ull * 2);  // rotated q bf16 [4096,8,64]
  __bf16* kbf  = (__bf16*)take(2097152ull * 2);  // rotated k bf16 [4096,8,64]
  __bf16* vbf  = (__bf16*)take(4194304ull * 2);  // v bf16 [4096,1024]
  __bf16* vT   = (__bf16*)take(4194304ull * 2);  // v^T bf16 [4][1024][1024]
  __bf16* abf  = (__bf16*)take(4194304ull * 2);  // attention out bf16 [4096,1024]

  cvt_kernel<<<4096, 256, 0, stream>>>(x, xbf, 1048576);
  cvtT_kernel<<<dim3(16, 32), 256, 0, stream>>>(w_t, wtT, 1024, 2048);
  cvtT_kernel<<<dim3(16, 16), 256, 0, stream>>>(w_o, woT, 1024, 1024);

  // t = x @ w_t   [4096 x 2048], K=1024
  gemm_bf16_kernel<<<dim3(64, 16), 256, 0, stream>>>(xbf, wtT, t, 4096, 2048, 1024);

  // masked rotation + bf16 repack
  rot_kernel<<<4096, 256, 0, stream>>>(t, rot, mask, qbf, kbf, vbf);

  // v^T per batch for contiguous attention staging
  transpose_v_kernel<<<dim3(16, 16, 4), 256, 0, stream>>>(vbf, vT);

  // Taylor attention
  attn_kernel<<<dim3(32, 16), 256, 0, stream>>>(qbf, kbf, vT, mask, abf);

  // out = att @ w_o   [4096 x 1024], K=1024
  gemm_bf16_kernel<<<dim3(64, 8), 256, 0, stream>>>(abf, woT, out, 4096, 1024, 1024);
}
